// MultiHeadAttention_89515708383514
// MI455X (gfx1250) — matrix operand order repaired and checked by reading, not compiled, not run on hardware
//
#include <hip/hip_runtime.h>
#include <hip/hip_bf16.h>

// MHA: B=4, S=2048, H=1024, NH=16, HD=64
#define B_ 4
#define S_ 2048
#define H_ 1024
#define NH_ 16
#define HD_ 64

typedef __attribute__((ext_vector_type(16))) _Float16 v16h;
typedef __attribute__((ext_vector_type(8)))  _Float16 v8h;
typedef __attribute__((ext_vector_type(8)))  float    v8f;

__device__ __forceinline__ v8f v8f_zero() {
    v8f v;
#pragma unroll
    for (int i = 0; i < 8; ++i) v[i] = 0.0f;
    return v;
}

__device__ __forceinline__ v8f wmma_f16(v16h a, v16h b, v8f c) {
    // D = A(16x32 f16) * B(32x16 f16) + C(16x16 f32)
    return __builtin_amdgcn_wmma_f32_16x16x32_f16(
        /*neg_a=*/false, a, /*neg_b=*/false, b,
        /*c_mod=*/(short)0, c, /*reuse_a=*/false, /*reuse_b=*/false);
}

// load 16 halves from two 16B-aligned chunks
__device__ __forceinline__ v16h ld_frag(const _Float16* p0, const _Float16* p1) {
    v16h r;
    ((uint4*)&r)[0] = *(const uint4*)p0;
    ((uint4*)&r)[1] = *(const uint4*)p1;
    return r;
}

// CDNA5 async copy: 16 bytes global -> LDS, tracked by ASYNCcnt.
// lds_addr is the workgroup-relative LDS byte offset (= low 32 bits of the
// flat address of a __shared__ object, per the LDS aperture rule).
__device__ __forceinline__ void async_ld_b128(unsigned lds_addr, const void* gaddr) {
    asm volatile("global_load_async_to_lds_b128 %0, %1, off"
                 :: "v"(lds_addr), "v"(gaddr)
                 : "memory");
}
#define WAIT_ASYNCCNT(n) asm volatile("s_wait_asynccnt " #n ::: "memory")

__device__ __forceinline__ unsigned lds_off(const void* p) {
    return (unsigned)(unsigned long long)p;   // low 32 bits of flat LDS address
}

// ---------------------------------------------------------------------------
// Kernel 1: fused QKV projection  y = x @ W^T + b   (M=8192, N=1024, K=1024)
// grid = (64, 8, 3); block = 256 (8 waves). Workgroup tile 128x128.
// z=0 -> Q (f16, [B,NH,S,HD]); z=1 -> K (same); z=2 -> V transposed [B,NH,HD,S]
// ---------------------------------------------------------------------------
__global__ __launch_bounds__(256)
void mha_qkv_gemm(const float* __restrict__ xq, const float* __restrict__ xk,
                  const float* __restrict__ xv,
                  const float* __restrict__ Wq, const float* __restrict__ bq,
                  const float* __restrict__ Wk, const float* __restrict__ bk,
                  const float* __restrict__ Wv, const float* __restrict__ bv,
                  _Float16* __restrict__ Qh, _Float16* __restrict__ Kh,
                  _Float16* __restrict__ Vt)
{
    __shared__ _Float16 Xs[128 * 32];   // 8 KB: X tile (rows m, cols k)
    __shared__ _Float16 Ws[128 * 32];   // 8 KB: W tile (rows n, cols k)

    const int z = blockIdx.z;
    const float* X    = (z == 0) ? xq : (z == 1) ? xk : xv;
    const float* W    = (z == 0) ? Wq : (z == 1) ? Wk : Wv;
    const float* bias = (z == 0) ? bq : (z == 1) ? bk : bv;

    const int mBase = blockIdx.x * 128;
    const int nBase = blockIdx.y * 128;
    const int tid   = threadIdx.x;
    const int wave  = tid >> 5;
    const int lane  = tid & 31;
    const int lhalf = lane >> 4;   // 0 or 1
    const int l16   = lane & 15;

    v8f acc[8];
#pragma unroll
    for (int j = 0; j < 8; ++j) acc[j] = v8f_zero();

    // cooperative load mapping: 256 threads cover 128 rows x 32 cols, 16 f32 each
    const int ldRow = tid >> 1;          // 0..127
    const int ldCol = (tid & 1) * 16;    // 0 or 16

    const float* srcX = X + (size_t)(mBase + ldRow) * H_ + ldCol;
    const float* srcW = W + (size_t)(nBase + ldRow) * H_ + ldCol;

    for (int k0 = 0; k0 < H_; k0 += 32) {
        // --- stage X tile (fp32 -> f16) ---
        {
            const float* src = srcX + k0;
            v8h h0, h1;
            float4 f0 = ((const float4*)src)[0];
            float4 f1 = ((const float4*)src)[1];
            float4 f2 = ((const float4*)src)[2];
            float4 f3 = ((const float4*)src)[3];
            h0[0]=(_Float16)f0.x; h0[1]=(_Float16)f0.y; h0[2]=(_Float16)f0.z; h0[3]=(_Float16)f0.w;
            h0[4]=(_Float16)f1.x; h0[5]=(_Float16)f1.y; h0[6]=(_Float16)f1.z; h0[7]=(_Float16)f1.w;
            h1[0]=(_Float16)f2.x; h1[1]=(_Float16)f2.y; h1[2]=(_Float16)f2.z; h1[3]=(_Float16)f2.w;
            h1[4]=(_Float16)f3.x; h1[5]=(_Float16)f3.y; h1[6]=(_Float16)f3.z; h1[7]=(_Float16)f3.w;
            *(v8h*)&Xs[ldRow * 32 + ldCol]     = h0;
            *(v8h*)&Xs[ldRow * 32 + ldCol + 8] = h1;
        }
        // --- stage W tile (fp32 -> f16) ---
        {
            const float* src = srcW + k0;
            v8h h0, h1;
            float4 f0 = ((const float4*)src)[0];
            float4 f1 = ((const float4*)src)[1];
            float4 f2 = ((const float4*)src)[2];
            float4 f3 = ((const float4*)src)[3];
            h0[0]=(_Float16)f0.x; h0[1]=(_Float16)f0.y; h0[2]=(_Float16)f0.z; h0[3]=(_Float16)f0.w;
            h0[4]=(_Float16)f1.x; h0[5]=(_Float16)f1.y; h0[6]=(_Float16)f1.z; h0[7]=(_Float16)f1.w;
            h1[0]=(_Float16)f2.x; h1[1]=(_Float16)f2.y; h1[2]=(_Float16)f2.z; h1[3]=(_Float16)f2.w;
            h1[4]=(_Float16)f3.x; h1[5]=(_Float16)f3.y; h1[6]=(_Float16)f3.z; h1[7]=(_Float16)f3.w;
            *(v8h*)&Ws[ldRow * 32 + ldCol]     = h0;
            *(v8h*)&Ws[ldRow * 32 + ldCol + 8] = h1;
        }
        // hint the next k-tile into cache while we compute (global_prefetch_b8)
        if (k0 + 32 < H_) {
            __builtin_prefetch(srcX + k0 + 32, 0, 0);
            __builtin_prefetch(srcW + k0 + 32, 0, 0);
        }
        __syncthreads();

        // A fragment: 16x32, lane=row(l16), K groups of 8 per half
        const _Float16* ab = &Xs[(wave * 16 + l16) * 32 + lhalf * 8];
        v16h a = ld_frag(ab, ab + 16);

#pragma unroll
        for (int j = 0; j < 8; ++j) {
            // B fragment: 32x16, lane=col, K groups of 8 per half
            const _Float16* bb = &Ws[(j * 16 + l16) * 32 + lhalf * 8];
            v16h b = ld_frag(bb, bb + 16);
            acc[j] = wmma_f16(a, b, acc[j]);
        }
        __syncthreads();
    }

    // Epilogue: add bias, cast f16, scatter to head-major layouts
    const int mRow0 = mBase + wave * 16 + lhalf * 8;
#pragma unroll
    for (int j = 0; j < 8; ++j) {
        const int n  = nBase + j * 16 + l16;
        const float bval = bias[n];
        const int h  = n >> 6;    // / HD
        const int d  = n & 63;    // % HD
#pragma unroll
        for (int r = 0; r < 8; ++r) {
            const int mrow = mRow0 + r;
            const int bb   = mrow >> 11;     // / S
            const int s    = mrow & 2047;    // % S
            const _Float16 val = (_Float16)(acc[j][r] + bval);
            if (z == 2) {
                Vt[(((size_t)bb * NH_ + h) * HD_ + d) * S_ + s] = val;
            } else if (z == 0) {
                Qh[(((size_t)bb * NH_ + h) * S_ + s) * HD_ + d] = val;
            } else {
                Kh[(((size_t)bb * NH_ + h) * S_ + s) * HD_ + d] = val;
            }
        }
    }
}

// ---------------------------------------------------------------------------
// Kernel 2: flash attention. grid = (S/128, NH, B); block = 256 (8 waves).
// Each wave owns 16 q-rows; loop over 64-key blocks with online softmax.
// K/V^T tiles are double-buffered in LDS via GLOBAL_LOAD_ASYNC_TO_LDS_B128.
// ---------------------------------------------------------------------------
__global__ __launch_bounds__(256)
void mha_flash_attn(const _Float16* __restrict__ Qh, const _Float16* __restrict__ Kh,
                    const _Float16* __restrict__ Vt, const float* __restrict__ mask,
                    float* __restrict__ out)
{
    __shared__ _Float16 Ks[2][64 * 64];     // 2 x 8 KB  K block  [key][d]
    __shared__ _Float16 Vs[2][64 * 64];     // 2 x 8 KB  V^T block [d][key]
    __shared__ _Float16 Ps[8][16 * 64];     // 16 KB per-wave P staging [row][key]

    const int qt   = blockIdx.x;
    const int h    = blockIdx.y;
    const int b    = blockIdx.z;
    const int tid  = threadIdx.x;
    const int wave = tid >> 5;
    const int lane = tid & 31;
    const int lhalf = lane >> 4;
    const int l16   = lane & 15;

    const _Float16* Qhead = Qh + ((size_t)b * NH_ + h) * S_ * HD_;
    const _Float16* Khead = Kh + ((size_t)b * NH_ + h) * S_ * HD_;
    const _Float16* Vhead = Vt + ((size_t)b * NH_ + h) * HD_ * S_;
    const float*    maskB = mask + (size_t)b * S_;

    const int q0 = qt * 128 + wave * 16;

    // Q A-fragments for this wave's 16 rows: 2 fragments cover d=0..63
    v16h qa[2];
    {
        const _Float16* qrow = Qhead + (size_t)(q0 + l16) * HD_ + lhalf * 8;
        qa[0] = ld_frag(qrow,      qrow + 16);
        qa[1] = ld_frag(qrow + 32, qrow + 48);
    }

    float mrow[8], lrow[8];
    v8f acc[4];
#pragma unroll
    for (int r = 0; r < 8; ++r) { mrow[r] = -1e30f; lrow[r] = 0.0f; }
#pragma unroll
    for (int j = 0; j < 4; ++j) acc[j] = v8f_zero();

    // cooperative K/V staging: 256 threads cover 64 rows x 64 halves, 16 each
    const int ldRow = tid >> 2;          // 0..63
    const int ldOff = (tid & 3) * 16;    // 0,16,32,48
    const unsigned ksLds0 = lds_off(&Ks[0][0]) + (unsigned)(ldRow * 64 + ldOff) * 2;
    const unsigned ksLds1 = lds_off(&Ks[1][0]) + (unsigned)(ldRow * 64 + ldOff) * 2;
    const unsigned vsLds0 = lds_off(&Vs[0][0]) + (unsigned)(ldRow * 64 + ldOff) * 2;
    const unsigned vsLds1 = lds_off(&Vs[1][0]) + (unsigned)(ldRow * 64 + ldOff) * 2;

    // issue this thread's 4 async b128 copies for key block kb into buffer buf
    auto issue_kv = [&](int buf, int kb) {
        const _Float16* sk = Khead + (size_t)(kb + ldRow) * HD_ + ldOff;
        const _Float16* sv = Vhead + (size_t)ldRow * S_ + kb + ldOff;
        const unsigned kd = buf ? ksLds1 : ksLds0;
        const unsigned vd = buf ? vsLds1 : vsLds0;
        async_ld_b128(kd,      sk);
        async_ld_b128(kd + 16, sk + 8);
        async_ld_b128(vd,      sv);
        async_ld_b128(vd + 16, sv + 8);
    };

    issue_kv(0, 0);   // prime buffer 0

    for (int kb = 0; kb < S_; kb += 64) {
        const int cur = (kb >> 6) & 1;
        const bool pref = (kb + 64 < S_);
        if (pref) issue_kv(cur ^ 1, kb + 64);   // prefetch next block
        // async loads complete in order per wave: <=4 outstanding => cur landed
        if (pref) { WAIT_ASYNCCNT(4); } else { WAIT_ASYNCCNT(0); }
        __syncthreads();

        const _Float16* Kc = &Ks[cur][0];
        const _Float16* Vc = &Vs[cur][0];

        // ---- scores: 16 q-rows x 64 keys ----
        v8f sf[4];
#pragma unroll
        for (int j = 0; j < 4; ++j) sf[j] = v8f_zero();
#pragma unroll
        for (int t = 0; t < 2; ++t) {
#pragma unroll
            for (int j = 0; j < 4; ++j) {
                const _Float16* bb = &Kc[(j * 16 + l16) * 64 + t * 32 + lhalf * 8];
                v16h kf = ld_frag(bb, bb + 16);
                sf[j] = wmma_f16(qa[t], kf, sf[j]);
            }
        }
        // scale 1/sqrt(64) + mask (mask depends only on key column)
#pragma unroll
        for (int j = 0; j < 4; ++j) {
            const float mv = maskB[kb + j * 16 + l16];
#pragma unroll
            for (int r = 0; r < 8; ++r) sf[j][r] = sf[j][r] * 0.125f + mv;
        }

        // ---- online softmax (rows live in 16-lane halves) ----
#pragma unroll
        for (int r = 0; r < 8; ++r) {
            float bm = fmaxf(fmaxf(sf[0][r], sf[1][r]), fmaxf(sf[2][r], sf[3][r]));
            bm = fmaxf(bm, __shfl_xor(bm, 1, 32));
            bm = fmaxf(bm, __shfl_xor(bm, 2, 32));
            bm = fmaxf(bm, __shfl_xor(bm, 4, 32));
            bm = fmaxf(bm, __shfl_xor(bm, 8, 32));
            const float mn   = fmaxf(mrow[r], bm);
            const float corr = __expf(mrow[r] - mn);
            float rs = 0.0f;
#pragma unroll
            for (int j = 0; j < 4; ++j) {
                const float p = __expf(sf[j][r] - mn);
                sf[j][r] = p;
                rs += p;
            }
            rs += __shfl_xor(rs, 1, 32);
            rs += __shfl_xor(rs, 2, 32);
            rs += __shfl_xor(rs, 4, 32);
            rs += __shfl_xor(rs, 8, 32);
            lrow[r] = lrow[r] * corr + rs;
            mrow[r] = mn;
#pragma unroll
            for (int j = 0; j < 4; ++j) acc[j][r] *= corr;
        }

        // ---- P (C-layout) -> LDS row-major, then reload as A-fragments ----
        _Float16* pw = &Ps[wave][0];
#pragma unroll
        for (int j = 0; j < 4; ++j)
#pragma unroll
            for (int r = 0; r < 8; ++r)
                pw[(lhalf * 8 + r) * 64 + j * 16 + l16] = (_Float16)sf[j][r];

        // ---- PV: acc += P(16x64) @ Vblk(64x64) ----
#pragma unroll
        for (int t = 0; t < 2; ++t) {
            const _Float16* pb = &pw[l16 * 64 + t * 32 + lhalf * 8];
            v16h pa = ld_frag(pb, pb + 16);
#pragma unroll
            for (int j = 0; j < 4; ++j) {
                const _Float16* bb = &Vc[(j * 16 + l16) * 64 + t * 32 + lhalf * 8];
                v16h vf = ld_frag(bb, bb + 16);
                acc[j] = wmma_f16(pa, vf, acc[j]);
            }
        }
        __syncthreads();
    }

    // ---- epilogue: normalize and write ctx to [B,S,H] fp32 ----
#pragma unroll
    for (int j = 0; j < 4; ++j) {
#pragma unroll
        for (int r = 0; r < 8; ++r) {
            const int row = lhalf * 8 + r;
            const int s   = q0 + row;
            const int d   = j * 16 + l16;
            out[((size_t)b * S_ + s) * H_ + h * HD_ + d] = acc[j][r] / lrow[r];
        }
    }
}

extern "C" void kernel_launch(void* const* d_in, const int* in_sizes, int n_in,
                              void* d_out, int out_size, void* d_ws, size_t ws_size,
                              hipStream_t stream) {
    (void)in_sizes; (void)n_in; (void)out_size; (void)ws_size;
    const float* xq   = (const float*)d_in[0];
    const float* xk   = (const float*)d_in[1];
    const float* xv   = (const float*)d_in[2];
    const float* mask = (const float*)d_in[3];
    const float* Wq   = (const float*)d_in[4];
    const float* bq   = (const float*)d_in[5];
    const float* Wk   = (const float*)d_in[6];
    const float* bk   = (const float*)d_in[7];
    const float* Wv   = (const float*)d_in[8];
    const float* bv   = (const float*)d_in[9];

    const size_t headElems = (size_t)B_ * NH_ * S_ * HD_;   // 8,388,608
    _Float16* Qh = (_Float16*)d_ws;
    _Float16* Kh = Qh + headElems;
    _Float16* Vt = Kh + headElems;

    dim3 g1(B_ * S_ / 128, H_ / 128, 3);   // (64, 8, 3)
    mha_qkv_gemm<<<g1, 256, 0, stream>>>(xq, xk, xv, Wq, bq, Wk, bk, Wv, bv,
                                         Qh, Kh, Vt);

    dim3 g2(S_ / 128, NH_, B_);            // (16, 16, 4)
    mha_flash_attn<<<g2, 256, 0, stream>>>(Qh, Kh, Vt, mask, (float*)d_out);
}
